// ssse_7816840479106
// MI455X (gfx1250) — compile-verified, matrix operand order checked by reading, not run on hardware
//
#include <hip/hip_runtime.h>
#include <hip/hip_bf16.h>

typedef __attribute__((ext_vector_type(16))) __bf16 v16bf;
typedef __attribute__((ext_vector_type(8)))  __bf16 v8bf;
typedef __attribute__((ext_vector_type(8)))  float  v8f;
typedef int vint4 __attribute__((vector_size(16)));   // b128 payload type

#define BATCH 64
#define HH 128
#define WWD 128
#define HWX (HH*WWD)      // 16384
#define EPSV 1e-5f

// gfx1250 async global->LDS copy path (guarded: falls back to sync copy)
#if defined(__gfx1250__) && __has_builtin(__builtin_amdgcn_global_load_async_to_lds_b128) && __has_builtin(__builtin_amdgcn_s_wait_asynccnt)
#define USE_ASYNC_LDS 1
#else
#define USE_ASYNC_LDS 0
#endif

__global__ void zero_f32(float* p, int n) {
    int i = blockIdx.x * blockDim.x + threadIdx.x;
    if (i < n) p[i] = 0.f;
}

// ---------------------------------------------------------------------------
// Weight pre-pack: f32 OIHW -> bf16 [COUT][KTP], K-order k = rs*CIN + ci,
// zero-padded to KTP = ceil32(CIN*R*R). Matches WMMA B-fragment consumption:
// each lane then loads two contiguous 16B bf16 vectors per 32-K chunk.
// ---------------------------------------------------------------------------
template<int R, int CIN, int COUT>
__global__ void prep_wgt(const float* __restrict__ w, __bf16* __restrict__ out) {
    constexpr int KT  = CIN * R * R;
    constexpr int KTP = ((KT + 31) / 32) * 32;
    const int tid = blockIdx.x * blockDim.x + threadIdx.x;
    if (tid >= COUT * KTP) return;
    const int n  = tid / KTP;
    const int kp = tid % KTP;
    float v = 0.f;
    if (kp < KT) {
        const int rs = kp / CIN;
        const int ci = kp % CIN;
        v = w[(n * CIN + ci) * (R * R) + rs];   // OIHW
    }
    out[tid] = (__bf16)v;
}

// ---------------------------------------------------------------------------
// conv1: 1x1, 16->32, NCHW f32 input (given layout), NHWC f32 output.
// A: scalar loads (K=16 zero-padded to 32); B: packed bf16 wide loads.
// ---------------------------------------------------------------------------
__global__ void __launch_bounds__(128)
conv1_wmma(const float* __restrict__ x, const __bf16* __restrict__ wb,
           float* __restrict__ out, float* __restrict__ stats) {
    constexpr int CIN = 16, COUT = 32, NT = 2;
    const int lane = threadIdx.x & 31;
    const int warp = threadIdx.x >> 5;
    const int wid  = blockIdx.x * 4 + warp;
    const int nt   = wid % NT;
    const int mt   = wid / NT;
    const int b    = mt / (HWX / 16);
    const int pixbase = (mt % (HWX / 16)) * 16;
    const int g    = lane >> 4;
    const int l15  = lane & 15;
    const int nbase = nt * 16;
    const int pix  = pixbase + l15;

    v16bf a;
#pragma unroll
    for (int e = 0; e < 16; ++e) {
        const int k = ((e & 8) << 1) + g * 8 + (e & 7);
        const float v = (k < CIN) ? x[((size_t)(b * CIN + k)) * HWX + pix] : 0.f;
        a[e] = (__bf16)v;
    }
    const __bf16* wp = wb + (nbase + l15) * 32 + g * 8;
    const v8bf blo = *(const v8bf*)wp;
    const v8bf bhi = *(const v8bf*)(wp + 16);
    const v16bf bb = __builtin_shufflevector(blo, bhi,
        0,1,2,3,4,5,6,7,8,9,10,11,12,13,14,15);
    v8f acc = {};
    acc = __builtin_amdgcn_wmma_f32_16x16x32_bf16(false, a, false, bb,
                                                  (short)0, acc, false, false);
    const int n = nbase + l15;
    float cs = 0.f, cq = 0.f;
#pragma unroll
    for (int r = 0; r < 8; ++r) {
        const float v = acc[r];
        out[((size_t)b * HWX + pixbase + r + 8 * g) * COUT + n] = v;
        cs += v; cq += v * v;
    }
    cs += __shfl_xor(cs, 16);
    cq += __shfl_xor(cq, 16);
    if (lane < 16) { atomicAdd(&stats[n], cs); atomicAdd(&stats[64 + n], cq); }
}

// ---------------------------------------------------------------------------
// Fused implicit-GEMM conv over NHWC pre-activations (BN scale/shift + ReLU
// applied while loading A). K = CIN*R*R (multiple of 32 for these layers),
// K-order rs*CIN+ci so every lane's 8-element K-segment is 8 contiguous
// channels -> two float4 loads. Packed bf16 weights are staged once per
// workgroup into LDS (async global->LDS on gfx1250), so per-chunk B
// fragments are two ds_load_b128s and the VMEM pipe is left to A.
// ---------------------------------------------------------------------------
template<int R, int CIN, int COUT>
__global__ void __launch_bounds__(128)
conv_wmma_nhwc(const float* __restrict__ in, const __bf16* __restrict__ wb,
               const float* __restrict__ ss_in,   // scale +0, shift +64
               float* __restrict__ out, float* __restrict__ stats) {
    constexpr int PAD = (R == 3) ? 1 : 0;
    constexpr int KT  = CIN * R * R;
    constexpr int NCH = KT / 32;
    constexpr int NT  = COUT / 16;
    constexpr int TOTW = COUT * KT;          // bf16 elements of packed weights
    constexpr int NV16 = TOTW / 8;           // 16-byte vectors

    __shared__ __align__(16) __bf16 wlds[TOTW];

    // stage packed weights into LDS, 16B per thread-iteration
    for (int i = threadIdx.x; i < NV16; i += 128) {
#if USE_ASYNC_LDS
        __builtin_amdgcn_global_load_async_to_lds_b128(
            (__attribute__((address_space(1))) vint4*)(wb + i * 8),
            (__attribute__((address_space(3))) vint4*)(wlds + i * 8), 0, 0);
#else
        *(v8bf*)(wlds + i * 8) = *(const v8bf*)(wb + i * 8);
#endif
    }
#if USE_ASYNC_LDS
    __builtin_amdgcn_s_wait_asynccnt(0);     // this wave's async writes done
#endif
    __syncthreads();                         // cross-wave visibility

    const int lane = threadIdx.x & 31;
    const int warp = threadIdx.x >> 5;
    const int wid  = blockIdx.x * 4 + warp;
    const int nt   = wid % NT;
    const int mt   = wid / NT;
    const int b    = mt / (HWX / 16);
    const int pixbase = (mt % (HWX / 16)) * 16;
    const int h0   = pixbase >> 7;
    const int w0   = pixbase & 127;
    const int g    = lane >> 4;
    const int l15  = lane & 15;
    const int nbase = nt * 16;

    v8f acc = {};
    for (int kc = 0; kc < NCH; ++kc) {
        const int kb = kc * 32;
        const __bf16* wp = wlds + (nbase + l15) * KT + kb + g * 8;
        const v8bf blo = *(const v8bf*)wp;
        const v8bf bhi = *(const v8bf*)(wp + 16);
        const v16bf bb = __builtin_shufflevector(blo, bhi,
            0,1,2,3,4,5,6,7,8,9,10,11,12,13,14,15);

        v16bf a;
#pragma unroll
        for (int s2 = 0; s2 < 2; ++s2) {
            const int ks = kb + 16 * s2 + g * 8;   // 8 contiguous channels
            int cb, hh, ww;
            if (R == 1) { cb = ks; hh = h0; ww = w0 + l15; }
            else {
                const int rs = ks / CIN;
                cb = ks - rs * CIN;
                hh = h0 + rs / 3 - PAD;
                ww = w0 + l15 + (rs % 3) - PAD;
            }
            float4 p0 = make_float4(0.f, 0.f, 0.f, 0.f);
            float4 p1 = make_float4(0.f, 0.f, 0.f, 0.f);
            if (R == 1 || (hh >= 0 && hh < HH && ww >= 0 && ww < WWD)) {
                const float* bp = in + ((size_t)b * HWX + hh * WWD + ww) * CIN + cb;
                p0 = *(const float4*)bp;
                p1 = *(const float4*)(bp + 4);
                if (kc + 1 < NCH) __builtin_prefetch((const void*)(bp + 32), 0, 0);
            }
            const float vv[8] = {p0.x, p0.y, p0.z, p0.w, p1.x, p1.y, p1.z, p1.w};
#pragma unroll
            for (int t = 0; t < 8; ++t) {
                float f = vv[t] * ss_in[cb + t] + ss_in[64 + cb + t];
                f = f > 0.f ? f : 0.f;
                a[s2 * 8 + t] = (__bf16)f;
            }
        }
        acc = __builtin_amdgcn_wmma_f32_16x16x32_bf16(false, a, false, bb,
                                                      (short)0, acc, false, false);
    }
    const int n = nbase + l15;
    float cs = 0.f, cq = 0.f;
#pragma unroll
    for (int r = 0; r < 8; ++r) {
        const float v = acc[r];
        out[((size_t)b * HWX + pixbase + r + 8 * g) * COUT + n] = v;
        cs += v; cq += v * v;
    }
    cs += __shfl_xor(cs, 16);
    cq += __shfl_xor(cq, 16);
    if (lane < 16) { atomicAdd(&stats[n], cs); atomicAdd(&stats[64 + n], cq); }
}

template<int COUT>
__global__ void bn_finalize(const float* __restrict__ stats,
                            const float* __restrict__ gamma,
                            const float* __restrict__ beta,
                            float* __restrict__ ss_out) {
    const int c = threadIdx.x;
    if (c < COUT) {
        const float inv_n = 1.f / (float)(BATCH * HWX);
        const float mean = stats[c] * inv_n;
        const float var  = stats[64 + c] * inv_n - mean * mean;
        const float sc   = gamma[c] * rsqrtf(var + EPSV);
        ss_out[c]      = sc;
        ss_out[64 + c] = beta[c] - mean * sc;
    }
}

// ---------------------------------------------------------------------------
// Per-pixel MLP on NHWC x2: tok(32) -> tanh(fc1,32) -> tanh(fc2,16) -> sum.
// 3 WMMAs/wave; LDS round-trip re-lays h1 as an A fragment. BN4+ReLU fused.
// ---------------------------------------------------------------------------
__global__ void __launch_bounds__(128)
token_mlp(const float* __restrict__ pre4, const float* __restrict__ ss4,
          const __bf16* __restrict__ fb1, const float* __restrict__ fc1b,
          const __bf16* __restrict__ fb2, const float* __restrict__ fc2b,
          float* __restrict__ y0sum) {
    __shared__ __bf16 lds[4][16][40];
    const int lane = threadIdx.x & 31;
    const int warp = threadIdx.x >> 5;
    const int wid  = blockIdx.x * 4 + warp;
    const int b    = wid / (HWX / 16);
    const int pixbase = (wid % (HWX / 16)) * 16;
    const int g   = lane >> 4;
    const int l15 = lane & 15;

    v16bf a;
    const float* bp = pre4 + ((size_t)b * HWX + pixbase + l15) * 32;
#pragma unroll
    for (int s2 = 0; s2 < 2; ++s2) {
        const int cb = 16 * s2 + g * 8;
        const float4 p0 = *(const float4*)(bp + cb);
        const float4 p1 = *(const float4*)(bp + cb + 4);
        const float vv[8] = {p0.x, p0.y, p0.z, p0.w, p1.x, p1.y, p1.z, p1.w};
#pragma unroll
        for (int t = 0; t < 8; ++t) {
            float f = vv[t] * ss4[cb + t] + ss4[64 + cb + t];
            f = f > 0.f ? f : 0.f;
            a[s2 * 8 + t] = (__bf16)f;
        }
    }
#pragma unroll
    for (int nt = 0; nt < 2; ++nt) {
        const __bf16* wp = fb1 + (nt * 16 + l15) * 32 + g * 8;
        const v8bf blo = *(const v8bf*)wp;
        const v8bf bhi = *(const v8bf*)(wp + 16);
        const v16bf bw = __builtin_shufflevector(blo, bhi,
            0,1,2,3,4,5,6,7,8,9,10,11,12,13,14,15);
        v8f d = {};
        d = __builtin_amdgcn_wmma_f32_16x16x32_bf16(false, a, false, bw,
                                                    (short)0, d, false, false);
        const int n = nt * 16 + l15;
        const float bias = fc1b[n];
#pragma unroll
        for (int r = 0; r < 8; ++r)
            lds[warp][r + 8 * g][n] = (__bf16)tanhf(d[r] + bias);
    }
    __syncthreads();
    v16bf a2;
#pragma unroll
    for (int e = 0; e < 16; ++e) {
        const int k = ((e & 8) << 1) + g * 8 + (e & 7);
        a2[e] = lds[warp][l15][k];
    }
    const __bf16* wp2 = fb2 + l15 * 32 + g * 8;
    const v8bf b2lo = *(const v8bf*)wp2;
    const v8bf b2hi = *(const v8bf*)(wp2 + 16);
    const v16bf b2 = __builtin_shufflevector(b2lo, b2hi,
        0,1,2,3,4,5,6,7,8,9,10,11,12,13,14,15);
    v8f d2 = {};
    d2 = __builtin_amdgcn_wmma_f32_16x16x32_bf16(false, a2, false, b2,
                                                 (short)0, d2, false, false);
    const float bias2 = fc2b[l15];
    float s = 0.f;
#pragma unroll
    for (int r = 0; r < 8; ++r) s += tanhf(d2[r] + bias2);
    s += __shfl_xor(s, 16);
    if (lane < 16) atomicAdd(&y0sum[b * 16 + l15], s);
}

// Per-batch head: y0, Cm, row norms, stable descending argsort, wi/vals/order.
__global__ void head(const float* __restrict__ y0sum, float* __restrict__ Cm,
                     int* __restrict__ idxbuf, float* __restrict__ out_wi,
                     float* __restrict__ out_vals, float* __restrict__ out_order) {
    const int b = threadIdx.x;
    if (b >= BATCH) return;
    float y0[16], wn[16];
    float n2 = 0.f;
    for (int i = 0; i < 16; ++i) {
        y0[i] = y0sum[b * 16 + i] * (1.f / (float)HWX);
        n2 += y0[i] * y0[i];
    }
    const float nrm = sqrtf(n2);
    for (int i = 0; i < 16; ++i) wn[i] = fabsf(y0[i]) * nrm;
    for (int i = 0; i < 16; ++i)
        for (int j = 0; j < 16; ++j)
            Cm[(b * 16 + i) * 16 + j] = y0[i] * y0[j];
    float sv[16]; int si[16];
    for (int i = 0; i < 16; ++i) { sv[i] = wn[i]; si[i] = i; }
    for (int i = 1; i < 16; ++i) {                // stable insertion, descending
        const float v = sv[i]; const int ix = si[i];
        int j = i - 1;
        while (j >= 0 && sv[j] < v) { sv[j + 1] = sv[j]; si[j + 1] = si[j]; --j; }
        sv[j + 1] = v; si[j + 1] = ix;
    }
    for (int i = 0; i < 16; ++i) {
        out_vals[b * 16 + i]  = sv[i];
        out_order[b * 16 + i] = (float)si[i];
        idxbuf[b * 16 + i]    = si[i];
    }
    for (int f = 0; f < 5; ++f)
        out_wi[f * BATCH + b] = (sv[3 * f] + sv[3 * f + 1] + sv[3 * f + 2]) * (1.f / 3.f);
}

// fi[f][b][c][p] = x[b][idx[b][3f+c]][p]   (pure bandwidth)
__global__ void gather_fi(const float* __restrict__ x, const int* __restrict__ idxbuf,
                          float* __restrict__ fi) {
    const size_t tid = (size_t)blockIdx.x * blockDim.x + threadIdx.x;
    const size_t total = (size_t)15 * BATCH * HWX;
    if (tid >= total) return;
    const int p = (int)(tid % HWX);
    int t = (int)(tid / HWX);
    const int c = t % 3; t /= 3;
    const int b = t % BATCH;
    const int f = t / BATCH;
    const int ch = idxbuf[b * 16 + f * 3 + c];
    fi[tid] = x[((size_t)(b * 16 + ch)) * HWX + p];
}

// cgh[b][i][p] = sum_j Cm[b][i][j] * x[b][j][p]  (4 FLOP/byte -> bandwidth bound)
__global__ void cgh_kernel(const float* __restrict__ x, const float* __restrict__ Cm,
                           float* __restrict__ cgh) {
    const int tid = blockIdx.x * blockDim.x + threadIdx.x;
    if (tid >= BATCH * HWX) return;
    const int b = tid / HWX, p = tid % HWX;
    float xv[16];
#pragma unroll
    for (int j = 0; j < 16; ++j) xv[j] = x[((size_t)(b * 16 + j)) * HWX + p];
#pragma unroll
    for (int i = 0; i < 16; ++i) {
        float s = 0.f;
#pragma unroll
        for (int j = 0; j < 16; ++j) s += Cm[(b * 16 + i) * 16 + j] * xv[j];
        cgh[((size_t)(b * 16 + i)) * HWX + p] = s;
    }
}

extern "C" void kernel_launch(void* const* d_in, const int* in_sizes, int n_in,
                              void* d_out, int out_size, void* d_ws, size_t ws_size,
                              hipStream_t stream) {
    const float* x   = (const float*)d_in[0];
    const float* w1  = (const float*)d_in[1];
    const float* g1  = (const float*)d_in[2];
    const float* be1 = (const float*)d_in[3];
    const float* w2  = (const float*)d_in[4];
    const float* g2  = (const float*)d_in[5];
    const float* be2 = (const float*)d_in[6];
    const float* w3  = (const float*)d_in[7];
    const float* g3  = (const float*)d_in[8];
    const float* be3 = (const float*)d_in[9];
    const float* w4  = (const float*)d_in[10];
    const float* g4  = (const float*)d_in[11];
    const float* be4 = (const float*)d_in[12];
    const float* f1w = (const float*)d_in[13];
    const float* f1b = (const float*)d_in[14];
    const float* f2w = (const float*)d_in[15];
    const float* f2b = (const float*)d_in[16];
    float* ws  = (float*)d_ws;
    float* out = (float*)d_out;

    // ws layout (floats): NHWC ping at 0 (32M), pong at 32M (64M), then smalls
    const size_t P1 = 0, P2 = 33554432ull, SMALL = 100663296ull;
    float* pre1 = ws + P1;
    float* pre2 = ws + P2;
    float* pre3 = ws + P1;   // pre1 dead after conv2
    float* pre4 = ws + P2;   // pre2 dead after conv3
    float* stats = ws + SMALL;          // 128
    float* ss1   = ws + SMALL + 128;
    float* ss2   = ws + SMALL + 256;
    float* ss3   = ws + SMALL + 384;
    float* ss4   = ws + SMALL + 512;
    float* y0sum = ws + SMALL + 640;    // 1024
    float* Cm    = ws + SMALL + 1664;   // 16384
    int*   idxbuf = (int*)(ws + SMALL + 18048);          // 1024 ints
    __bf16* wbase = (__bf16*)(ws + SMALL + 19072);       // packed weights
    __bf16* wb1 = wbase;          // 32*32   = 1024
    __bf16* wb2 = wbase + 1024;   // 64*32   = 2048
    __bf16* wb3 = wbase + 3072;   // 32*576  = 18432
    __bf16* wb4 = wbase + 21504;  // 32*288  = 9216
    __bf16* fb1 = wbase + 30720;  // 32*32   = 1024
    __bf16* fb2 = wbase + 31744;  // 16*32   = 512

    // d_out: fi(15728640) | wi(320) | vals(1024) | order(1024) | cgh(16777216)
    float* out_fi    = out;
    float* out_wi    = out + 15728640;
    float* out_vals  = out + 15728960;
    float* out_order = out + 15729984;
    float* out_cgh   = out + 15731008;

    const int MT = BATCH * HWX / 16;    // 65536 pixel tiles

    zero_f32<<<1, 128, 0, stream>>>(stats, 128);
    zero_f32<<<8, 128, 0, stream>>>(y0sum, 1024);

    prep_wgt<1, 16, 32><<<8,   128, 0, stream>>>(w1,  wb1);
    prep_wgt<1, 32, 64><<<16,  128, 0, stream>>>(w2,  wb2);
    prep_wgt<3, 64, 32><<<144, 128, 0, stream>>>(w3,  wb3);
    prep_wgt<3, 32, 32><<<72,  128, 0, stream>>>(w4,  wb4);
    prep_wgt<1, 32, 32><<<8,   128, 0, stream>>>(f1w, fb1);
    prep_wgt<1, 32, 16><<<4,   128, 0, stream>>>(f2w, fb2);

    conv1_wmma<<<MT * 2 / 4, 128, 0, stream>>>(x, wb1, pre1, stats);
    bn_finalize<32><<<1, 64, 0, stream>>>(stats, g1, be1, ss1);
    zero_f32<<<1, 128, 0, stream>>>(stats, 128);

    conv_wmma_nhwc<1, 32, 64><<<MT * 4 / 4, 128, 0, stream>>>(pre1, wb2, ss1, pre2, stats);
    bn_finalize<64><<<1, 64, 0, stream>>>(stats, g2, be2, ss2);
    zero_f32<<<1, 128, 0, stream>>>(stats, 128);

    conv_wmma_nhwc<3, 64, 32><<<MT * 2 / 4, 128, 0, stream>>>(pre2, wb3, ss2, pre3, stats);
    bn_finalize<32><<<1, 64, 0, stream>>>(stats, g3, be3, ss3);
    zero_f32<<<1, 128, 0, stream>>>(stats, 128);

    conv_wmma_nhwc<3, 32, 32><<<MT * 2 / 4, 128, 0, stream>>>(pre3, wb4, ss3, pre4, stats);
    bn_finalize<32><<<1, 64, 0, stream>>>(stats, g4, be4, ss4);

    token_mlp<<<MT / 4, 128, 0, stream>>>(pre4, ss4, fb1, f1b, fb2, f2b, y0sum);
    head<<<1, 64, 0, stream>>>(y0sum, Cm, idxbuf, out_wi, out_vals, out_order);

    const size_t fitotal = (size_t)15 * BATCH * HWX;
    gather_fi<<<(unsigned)((fitotal + 255) / 256), 256, 0, stream>>>(x, idxbuf, out_fi);
    cgh_kernel<<<(BATCH * HWX + 255) / 256, 256, 0, stream>>>(x, Cm, out_cgh);
}
